// RGCN_30193620091085
// MI455X (gfx1250) — matrix operand order repaired and checked by reading, not compiled, not run on hardware
//
#include <hip/hip_runtime.h>
#include <hip/hip_bf16.h>
#include <stdint.h>

// ---------------------------------------------------------------------------
// RGCN (2 layers) for MI455X / gfx1250.
// Memory-bound: agg intermediate kept bf16 (102.4MB, L2-resident), scatter via
// global_atomic_pk_add_bf16, dense math via v_wmma bf16 with weights staged in
// LDS by the Tensor Data Mover (tensor_load_to_lds + s_wait_tensorcnt).
// ---------------------------------------------------------------------------

typedef float        v8f   __attribute__((ext_vector_type(8)));
typedef __bf16       v16bf __attribute__((ext_vector_type(16)));
typedef unsigned int v4u   __attribute__((ext_vector_type(4)));
typedef int          v8i   __attribute__((ext_vector_type(8)));
typedef int          v4i   __attribute__((ext_vector_type(4)));

union V16 {
    uint4 u4[2];
    v16bf v;
};

__device__ __forceinline__ unsigned short f2bf(float f) {
    unsigned u = __builtin_bit_cast(unsigned, f);
    u += 0x7FFFu + ((u >> 16) & 1u);          // round-to-nearest-even
    return (unsigned short)(u >> 16);
}

__device__ __forceinline__ float bf2f(unsigned short h) {
    return __builtin_bit_cast(float, (unsigned)h << 16);
}

__device__ __forceinline__ unsigned scale_pack2(unsigned packed, float s) {
    float f0 = bf2f((unsigned short)(packed & 0xFFFFu)) * s;
    float f1 = bf2f((unsigned short)(packed >> 16)) * s;
    return (unsigned)f2bf(f0) | ((unsigned)f2bf(f1) << 16);
}

// ---------------- f32 -> bf16 conversion (2 elems / thread) ----------------
__global__ void cvt_bf16_kernel(const float2* __restrict__ x,
                                unsigned* __restrict__ out, int n2) {
    int i = blockIdx.x * blockDim.x + threadIdx.x;
    if (i >= n2) return;
    float2 v = x[i];
    out[i] = (unsigned)f2bf(v.x) | ((unsigned)f2bf(v.y) << 16);
}

// -------- transpose W [R][K][F] (+root [K][F]) -> bf16 Wt [R+1][F][K] ------
__global__ void prep_w_kernel(const float* __restrict__ W,
                              const float* __restrict__ root,
                              unsigned short* __restrict__ Wt,
                              int K, int F) {
    int idx = blockIdx.x * blockDim.x + threadIdx.x;
    int total = 9 * F * K;
    if (idx >= total) return;
    int seg = idx / (F * K);
    int rem = idx - seg * F * K;
    int col = rem / K;
    int k   = rem - col * K;
    float v = (seg < 8) ? W[((size_t)seg * K + k) * F + col]
                        : root[(size_t)k * F + col];
    Wt[idx] = f2bf(v);
}

// ---------------- per-(dst,rel) edge count (once, reused) ------------------
__global__ void count_kernel(const int* __restrict__ dst,
                             const int* __restrict__ et,
                             int* __restrict__ cnt, int E) {
    int e = blockIdx.x * blockDim.x + threadIdx.x;
    if (e >= E) return;
    atomicAdd(&cnt[dst[e] * 8 + et[e]], 1);
}

// --- scatter: agg[dst*8+rel] += x[src]/cnt   (one wave32 per edge) ---------
__global__ void scatter_kernel(const unsigned short* __restrict__ xb, // bf16 [N][128]
                               const int* __restrict__ src,
                               const int* __restrict__ dst,
                               const int* __restrict__ et,
                               const int* __restrict__ cnt,
                               unsigned short* __restrict__ agg,      // bf16 [N*8][128]
                               int E) {
    int wave = (blockIdx.x * blockDim.x + threadIdx.x) >> 5;
    int lane = threadIdx.x & 31;
    if (wave >= E) return;
    int e = wave;
    int s = src[e];
    int seg = dst[e] * 8 + et[e];
    float scale = 1.0f / (float)cnt[seg];
    const uint2* xp = reinterpret_cast<const uint2*>(xb + (size_t)s * 128) + lane;
    uint2 v = *xp;
    unsigned p0 = scale_pack2(v.x, scale);
    unsigned p1 = scale_pack2(v.y, scale);
    uint64_t addr = (uint64_t)(agg + (size_t)seg * 128 + lane * 4);
    asm volatile("global_atomic_pk_add_bf16 %0, %1, off"
                 :: "v"(addr), "v"(p0) : "memory");
    asm volatile("global_atomic_pk_add_bf16 %0, %1, off"
                 :: "v"(addr + 4), "v"(p1) : "memory");
}

// ------------------------- WMMA bf16 GEMM ----------------------------------
// out[n,:] = act( sum_{seg<8} agg[n*8+seg,:] @ Wt[seg] + xroot[n,:] @ Wt[8] + bias )
// Block = 256 threads = 8 waves; each wave owns TWO 16-row strips (M=256/block).
// Full Wt [9][FOUT][128] staged into LDS via TDM; A and B double-buffered so
// global/LDS latency overlaps the XDL pipe; VGPR allocation stays under 256.
template <int FOUT, bool RELU, bool OUT_BF16>
__global__ __launch_bounds__(256, 2) void rgcn_gemm_kernel(
    const unsigned short* __restrict__ agg,    // bf16 [N*8][128] (row-major: row*1024 + seg*128)
    const unsigned short* __restrict__ xroot,  // bf16 [N][128]
    const unsigned short* __restrict__ Wt,     // bf16 [9][FOUT][128]
    const float* __restrict__ bias,            // [FOUT]
    void* __restrict__ outv,                   // [N][FOUT]
    int nrows) {
    constexpr int NT = FOUT / 16;
    constexpr unsigned WT_ELEMS = 9u * FOUT * 128;   // bf16 elements
    extern __shared__ unsigned short ldsW[];

    // ---- stage Wt -> LDS -------------------------------------------------
#if __has_builtin(__builtin_amdgcn_tensor_load_to_lds) && __has_builtin(__builtin_amdgcn_s_wait_tensorcnt)
    if (threadIdx.x == 0) {
        // 2-D D# descriptor: 8-byte elements, tile = tensor = [9][FOUT*32]
        const unsigned td0   = FOUT * 32u;            // dim0 length (8B units)
        const unsigned td1   = 9u;
        const uint64_t gaddr = (uint64_t)Wt;
        v4u g0;
        g0[0] = 1u;                                   // count=1 (valid descriptor)
        g0[1] = 0u;                                   // lds_addr = 0 (dynamic LDS base)
        g0[2] = (unsigned)(gaddr & 0xFFFFFFFFu);      // global_addr[31:0]
        g0[3] = (unsigned)((gaddr >> 32) & 0x01FFFFFFu) | (2u << 30); // addr[56:32] | type=2
        v8i g1;
        g1[0] = (int)(3u << 16);                      // workgroup_mask=0, data_size=3 (8B)
        g1[1] = (int)((td0 & 0xFFFFu) << 16);         // tensor_dim0[15:0]
        g1[2] = (int)((td0 >> 16) | ((td1 & 0xFFFFu) << 16)); // td0[31:16] | td1[15:0]
        g1[3] = (int)((td1 >> 16) | (td0 << 16));     // td1[31:16] | tile_dim0
        g1[4] = (int)td1;                             // tile_dim1 (tile_dim2 = 0)
        g1[5] = (int)td0;                             // tensor_dim0_stride[31:0]
        g1[6] = 0;                                    // stride0[47:32] | stride1[15:0]
        g1[7] = 0;                                    // stride1[47:16]
        v4i g2 = {0, 0, 0, 0};
        v4i g3 = {0, 0, 0, 0};
        v8i g4 = {0, 0, 0, 0, 0, 0, 0, 0};            // unused group (6-arg toolchain form)
        __builtin_amdgcn_tensor_load_to_lds(g0, g1, g2, g3, g4, 0);
        __builtin_amdgcn_s_wait_tensorcnt(0);
    }
#else
    {
        const uint4* s4 = reinterpret_cast<const uint4*>(Wt);
        uint4* d4 = reinterpret_cast<uint4*>(ldsW);
        for (unsigned i = threadIdx.x; i < WT_ELEMS / 8; i += 256) d4[i] = s4[i];
    }
#endif
    __syncthreads();

    const int wave = threadIdx.x >> 5;
    const int lane = threadIdx.x & 31;
    const int half = lane >> 4;                 // 0: lanes 0-15, 1: lanes 16-31
    const int l15  = lane & 15;
    const int blockBase = blockIdx.x * 256;

    int r0 = blockBase + wave * 16 + l15;
    int r1 = r0 + 128;
    int r0c = r0 < nrows ? r0 : nrows - 1;      // clamp once; stores are guarded
    int r1c = r1 < nrows ? r1 : nrows - 1;

    // A base pointers with the lane's K-half folded in.
    const unsigned short* aB0 = agg + (size_t)r0c * 1024 + half * 8;
    const unsigned short* aB1 = agg + (size_t)r1c * 1024 + half * 8;
    const unsigned short* xB0 = xroot + (size_t)r0c * 128 + half * 8;
    const unsigned short* xB1 = xroot + (size_t)r1c * 128 + half * 8;

    v8f c[2][NT] = {};

    // A 16x32 bf16 layout: elems 0-7 = K k0..k0+7, elems 8-15 = K k0+16..k0+23
    auto ldIter = [&](int it, V16& A0, V16& A1) {
        if (it < 32) {
            int off = it * 32;
            A0.u4[0] = *reinterpret_cast<const uint4*>(aB0 + off);
            A0.u4[1] = *reinterpret_cast<const uint4*>(aB0 + off + 16);
            A1.u4[0] = *reinterpret_cast<const uint4*>(aB1 + off);
            A1.u4[1] = *reinterpret_cast<const uint4*>(aB1 + off + 16);
        } else {
            int off = (it & 3) * 32;
            A0.u4[0] = *reinterpret_cast<const uint4*>(xB0 + off);
            A0.u4[1] = *reinterpret_cast<const uint4*>(xB0 + off + 16);
            A1.u4[0] = *reinterpret_cast<const uint4*>(xB1 + off);
            A1.u4[1] = *reinterpret_cast<const uint4*>(xB1 + off + 16);
        }
    };

    // B 32x16 bf16 layout: elems 0-7 = K k0+half*8..+7, elems 8-15 = K k0+16+half*8..+7 for col = lane&15.
    // seg = it>>2, ks = it&3 holds for all 36 k-steps (root is seg 8 in ldsW).
    auto ldB = [&](const unsigned short* bSeg, int nt, V16& B) {
        B.u4[0] = *reinterpret_cast<const uint4*>(bSeg + nt * 2048);
        B.u4[1] = *reinterpret_cast<const uint4*>(bSeg + nt * 2048 + 16);
    };

    auto mmIter = [&](int it, const V16& A0, const V16& A1) {
        const int seg = it >> 2;
        const int ks  = it & 3;
        const unsigned short* bSeg =
            ldsW + (unsigned)(seg * FOUT + l15) * 128 + ks * 32 + half * 8;
        V16 b0, b1;                               // B ping-pong: load nt+1 under nt's WMMAs
        ldB(bSeg, 0, b0);
#pragma unroll
        for (int nt = 0; nt < NT; ++nt) {
            V16& cur = (nt & 1) ? b1 : b0;
            V16& nxt = (nt & 1) ? b0 : b1;
            if (nt + 1 < NT) ldB(bSeg, nt + 1, nxt);
            c[0][nt] = __builtin_amdgcn_wmma_f32_16x16x32_bf16(
                false, A0.v, false, cur.v, (short)0, c[0][nt], false, false);
            c[1][nt] = __builtin_amdgcn_wmma_f32_16x16x32_bf16(
                false, A1.v, false, cur.v, (short)0, c[1][nt], false, false);
        }
    };

    // 36 k-steps, ping-pong A double buffer (no register copies).
    V16 pa0, pa1, pb0, pb1;
    ldIter(0, pa0, pa1);
    for (int it = 0; it < 36; it += 2) {
        ldIter(it + 1, pb0, pb1);
        mmIter(it, pa0, pa1);
        if (it + 2 < 36) ldIter(it + 2, pa0, pa1);
        mmIter(it + 1, pb0, pb1);
    }

    // Epilogue. C/D layout: VGPR i -> M = i + half*8, N = lane&15.
#pragma unroll
    for (int t = 0; t < 2; ++t) {
        int rowB = blockBase + wave * 16 + t * 128 + half * 8;
#pragma unroll
        for (int nt = 0; nt < NT; ++nt) {
            int col = nt * 16 + l15;
            float bv = bias[col];
#pragma unroll
            for (int i = 0; i < 8; ++i) {
                int row = rowB + i;
                if (row < nrows) {
                    float v = c[t][nt][i] + bv;
                    if (RELU) v = fmaxf(v, 0.0f);
                    if (OUT_BF16)
                        ((unsigned short*)outv)[(size_t)row * FOUT + col] = f2bf(v);
                    else
                        ((float*)outv)[(size_t)row * FOUT + col] = v;
                }
            }
        }
    }
}

// ---------------------------------------------------------------------------
extern "C" void kernel_launch(void* const* d_in, const int* in_sizes, int n_in,
                              void* d_out, int out_size, void* d_ws, size_t ws_size,
                              hipStream_t stream) {
    const float* x     = (const float*)d_in[0];
    const int*   eidx  = (const int*)d_in[1];
    const int*   etype = (const int*)d_in[2];
    const float* W1    = (const float*)d_in[3];
    const float* root1 = (const float*)d_in[4];
    const float* b1    = (const float*)d_in[5];
    const float* W2    = (const float*)d_in[6];
    const float* root2 = (const float*)d_in[7];
    const float* b2    = (const float*)d_in[8];

    const int N = in_sizes[0] / 128;   // Fin = 128
    const int E = in_sizes[1] / 2;
    const int* src = eidx;
    const int* dst = eidx + E;

    auto al = [](size_t v) { return (v + 255) & ~(size_t)255; };
    char* ws = (char*)d_ws;
    size_t off = 0;
    unsigned short* xb  = (unsigned short*)(ws + off); off += al((size_t)N * 128 * 2);
    unsigned short* hb  = (unsigned short*)(ws + off); off += al((size_t)N * 128 * 2);
    int*            cnt = (int*)           (ws + off); off += al((size_t)N * 8 * 4);
    unsigned short* Wt1 = (unsigned short*)(ws + off); off += al((size_t)9 * 128 * 128 * 2);
    unsigned short* Wt2 = (unsigned short*)(ws + off); off += al((size_t)9 * 64 * 128 * 2);
    unsigned short* agg = (unsigned short*)(ws + off);
    const size_t aggBytes = (size_t)N * 8 * 128 * 2;   // 102.4 MB (bf16, L2-resident)

    const int T = 256;
    (void)hipMemsetAsync(cnt, 0, (size_t)N * 8 * 4, stream);
    {
        int n2 = N * 64;
        cvt_bf16_kernel<<<(n2 + T - 1) / T, T, 0, stream>>>((const float2*)x, (unsigned*)xb, n2);
    }
    {
        int tot1 = 9 * 128 * 128;
        prep_w_kernel<<<(tot1 + T - 1) / T, T, 0, stream>>>(W1, root1, Wt1, 128, 128);
        int tot2 = 9 * 64 * 128;
        prep_w_kernel<<<(tot2 + T - 1) / T, T, 0, stream>>>(W2, root2, Wt2, 128, 64);
    }
    count_kernel<<<(E + T - 1) / T, T, 0, stream>>>(dst, etype, cnt, E);

    const int gemmBlocks = (N + 255) / 256;
    const int scatBlocks = (int)(((size_t)E * 32 + T - 1) / T);
    const size_t lds1 = (size_t)9 * 128 * 128 * 2;   // 288 KB  (<= 320 KB/WGP)
    const size_t lds2 = (size_t)9 * 64 * 128 * 2;    // 144 KB

    // ---- layer 1 ----
    (void)hipMemsetAsync(agg, 0, aggBytes, stream);
    scatter_kernel<<<scatBlocks, T, 0, stream>>>(xb, src, dst, etype, cnt, agg, E);
    rgcn_gemm_kernel<128, true, true><<<gemmBlocks, T, lds1, stream>>>(
        agg, xb, Wt1, b1, hb, N);

    // ---- layer 2 ----
    (void)hipMemsetAsync(agg, 0, aggBytes, stream);
    scatter_kernel<<<scatBlocks, T, 0, stream>>>(hb, src, dst, etype, cnt, agg, E);
    rgcn_gemm_kernel<64, false, false><<<gemmBlocks, T, lds2, stream>>>(
        agg, hb, Wt2, b2, d_out, N);
}
